// Seq2SeqRNN_All_68667937129210
// MI455X (gfx1250) — compile-verified, matrix operand order checked by reading, not run on hardware
//
#include <hip/hip_runtime.h>
#include <hip/hip_bf16.h>

// ---------------------------------------------------------------------------
// Seq2Seq GRU encoder/decoder with attention + copy mechanism, for MI455X.
// All GEMMs (X @ W.T, NT layout) run through v_wmma_f32_16x16x32_f16 with
// f32 accumulation; weights are converted to f16 once (emb_dec f16 = 49MB,
// fits the 192MB L2 so the 48 decoder vocab projections stream from L2).
// The sequential encoder GRU recurrence runs as a persistent fused kernel:
// one workgroup per direction keeps h (f32+f16) and the gate buffer in the
// 320KB CDNA5 LDS (288KB used), WMMA A-fragments come from ds_load_b128,
// eliminating ~512 tiny dependent kernel launches.
// ---------------------------------------------------------------------------

typedef _Float16 f16_t;
typedef __attribute__((ext_vector_type(16))) _Float16 v16h;
typedef __attribute__((ext_vector_type(8)))  float    v8f;

#define SL     64
#define BS     64
#define OUT_SL 48
#define NH     256
#define EME    512
#define EMD    768
#define VSZ    32000
#define VC     (VSZ + SL)      // 32064

// ---------------------------------------------------------------------------
// Fragment loader for v_wmma_*_16x16x32_f16 (wave32):
//   lane = half*16 + r ; element e (0..15) holds K = k0 + half*8 + (e&7) + (e>>3)*16
// -> two contiguous 16-byte loads per lane (K offsets +0..7 and +16..23).
// Works for global or LDS pointers (ds_load_b128 when p is shared).
// ---------------------------------------------------------------------------
__device__ __forceinline__ v16h load_frag16(const f16_t* p) {
    union { uint4 q; f16_t h[8]; } lo, hi;
    lo.q = *(const uint4*)p;
    hi.q = *(const uint4*)(p + 16);
    v16h f;
#pragma unroll
    for (int i = 0; i < 8; ++i) { f[i] = lo.h[i]; f[i + 8] = hi.h[i]; }
    return f;
}

__device__ __forceinline__ float sigmoidf_(float x) { return 1.0f / (1.0f + expf(-x)); }

// ---------------------------------------------------------------------------
// Generic NT GEMM:  C[M,N] = act( A[M,K] * B[N,K]^T + bias[N] )
// One wave per 16x16 tile; block = 8 waves covering 128 consecutive columns.
// Requires M%16==0, N%128==0, K%32==0.
// ---------------------------------------------------------------------------
__global__ __launch_bounds__(256)
void gemm_nt_kernel(const f16_t* __restrict__ A, int lda,
                    const f16_t* __restrict__ B, int ldb,
                    float* __restrict__ C, int ldc,
                    f16_t* __restrict__ C16, int ldc16,
                    const float* __restrict__ bias,
                    int K, int act)   // act: 0 = none, 1 = tanh
{
    const int lane = threadIdx.x & 31;
    const int wave = threadIdx.x >> 5;
    const int n0   = (blockIdx.x * 8 + wave) * 16;
    const int m0   = blockIdx.y * 16;
    const int half = lane >> 4;
    const int r    = lane & 15;

    const f16_t* arow = A + (size_t)(m0 + r) * lda + half * 8;
    const f16_t* brow = B + (size_t)(n0 + r) * ldb + half * 8;

    v8f acc = {};
    for (int k0 = 0; k0 < K; k0 += 32) {
        if (k0 + 32 < K) {                       // uniform; hint next B/A tiles
            __builtin_prefetch(brow + k0 + 32, 0, 3);
            __builtin_prefetch(arow + k0 + 32, 0, 3);
        }
        v16h a = load_frag16(arow + k0);
        v16h b = load_frag16(brow + k0);
        acc = __builtin_amdgcn_wmma_f32_16x16x32_f16(
            /*neg_a=*/false, a, /*neg_b=*/false, b,
            /*c_mod=*/(short)0, acc, /*reuse_a=*/false, /*reuse_b=*/false);
    }

    // C/D layout: lane (half*16+n), VGPR v -> element (m0 + half*8 + v, n0 + n)
    const int n     = n0 + r;
    const int mbase = m0 + half * 8;
    const float bv  = bias ? bias[n] : 0.0f;
#pragma unroll
    for (int v = 0; v < 8; ++v) {
        float x = acc[v] + bv;
        if (act == 1) x = tanhf(x);
        C[(size_t)(mbase + v) * ldc + n] = x;
        if (C16) C16[(size_t)(mbase + v) * ldc16 + n] = (f16_t)x;
    }
}

// ---------------------------------------------------------------------------
// Persistent fused encoder GRU direction kernel.
// grid.x = 2 (directions), 1024 threads (32 waves), dynamic LDS:
//   h32s  64x256 f32 (64KB) | h16s 64x256 f16 (32KB) | ghs 64x768 f32 (192KB)
// Per step: gh = h16s @ Whh^T + bhh via WMMA (A from LDS, B from global/L2),
// then fused GRU gates updating LDS-resident state + streaming layer outputs.
// ---------------------------------------------------------------------------
#define FUSED_LDS_BYTES (64*256*4 + 64*256*2 + 64*768*4)   // 294912 < 320KB

__global__ __launch_bounds__(1024)
void gru_dir_fused_kernel(const float* __restrict__ giF, const float* __restrict__ giB,
                          const f16_t* __restrict__ WhhL,   // layer base: 2 x (768,256)
                          const float* __restrict__ bhhL,   // layer base: 2 x 768
                          float* __restrict__ hfinL,        // layer base: 2 x (64,256)
                          float* __restrict__ y32,          // nullable, (s,b,512)
                          f16_t* __restrict__ y16)          // (s,b,512)
{
    extern __shared__ uint4 smem_u4[];
    char*  smem = (char*)smem_u4;
    float* h32s = (float*)smem;                              // 64*256
    f16_t* h16s = (f16_t*)(smem + 64*256*4);                 // 64*256
    float* ghs  = (float*)(smem + 64*256*4 + 64*256*2);      // 64*768

    const int d    = blockIdx.x;
    const int tid  = threadIdx.x;
    const int lane = tid & 31;
    const int wave = tid >> 5;
    const int half = lane >> 4;
    const int r    = lane & 15;

    const float* gi  = d ? giB : giF;
    const f16_t* Whh = WhhL + (size_t)d * 768 * 256;
    const float* bhh = bhhL + d * 768;

    for (int i = tid; i < 64 * 256; i += 1024) { h32s[i] = 0.0f; h16s[i] = (f16_t)0.0f; }
    __syncthreads();

    for (int step = 0; step < SL; ++step) {
        const int s = d ? (SL - 1 - step) : step;

        // ---- gh = h @ Whh^T + bhh : (64,256)@(256,768) = 4 x 48 tiles ----
        for (int tile = wave; tile < 4 * 48; tile += 32) {   // exactly 6 per wave
            const int tm = tile & 3;
            const int tn = tile >> 2;
            const int m0 = tm * 16, n0 = tn * 16;
            const f16_t* arow = h16s + (m0 + r) * 256 + half * 8;   // LDS
            const f16_t* brow = Whh + (size_t)(n0 + r) * 256 + half * 8;
            v8f acc = {};
#pragma unroll
            for (int k0 = 0; k0 < 256; k0 += 32) {
                v16h a = load_frag16(arow + k0);    // ds_load_b128 x2
                v16h b = load_frag16(brow + k0);    // global_load_b128 x2 (L2)
                acc = __builtin_amdgcn_wmma_f32_16x16x32_f16(
                    false, a, false, b, (short)0, acc, false, false);
            }
            const int n     = n0 + r;
            const int mbase = m0 + half * 8;
            const float bv  = bhh[n];
#pragma unroll
            for (int v = 0; v < 8; ++v)
                ghs[(mbase + v) * 768 + n] = acc[v] + bv;
        }
        __syncthreads();

        // ---- fused GRU pointwise; 16384 elems over 1024 threads ----
        const float* gis = gi + (size_t)s * 64 * 768;
        for (int i = tid; i < 64 * 256; i += 1024) {
            const int b = i >> 8, j = i & 255;
            const float* gib = gis + b * 768;
            const float* ghb = ghs + b * 768;
            float rg = sigmoidf_(gib[j]       + ghb[j]);
            float zg = sigmoidf_(gib[256 + j] + ghb[256 + j]);
            float ng = tanhf(    gib[512 + j] + rg * ghb[512 + j]);
            float h  = (1.0f - zg) * ng + zg * h32s[i];
            h32s[i] = h;
            h16s[i] = (f16_t)h;
            const size_t yo = (size_t)(s * 64 + b) * 512 + d * 256 + j;
            y16[yo] = (f16_t)h;
            if (y32) y32[yo] = h;
        }
        __syncthreads();
    }

    float* hfin = hfinL + (size_t)d * 64 * 256;
    for (int i = tid; i < 64 * 256; i += 1024) hfin[i] = h32s[i];
}

// ---------------------------------------------------------------------------
// Elementwise / helper kernels
// ---------------------------------------------------------------------------
__global__ void cvt_f16_kernel(f16_t* dst, const float* src, int n) {
    int i = blockIdx.x * 256 + threadIdx.x;
    if (i < n) dst[i] = (f16_t)src[i];
}

// dst[r*cols+c] = src[c*rows+r]   (transpose + f16 convert)
__global__ void cvtT_f16_kernel(f16_t* dst, const float* src, int rows, int cols) {
    int i = blockIdx.x * 256 + threadIdx.x;
    if (i >= rows * cols) return;
    int r = i / cols, c = i % cols;
    dst[i] = (f16_t)src[(size_t)c * rows + r];
}

// x = (f16) emb_enc[inp]   -> (SL*BS, EME)
__global__ void embed_enc_kernel(f16_t* dst, const float* emb, const int* inp) {
    int i = blockIdx.x * 256 + threadIdx.x;      // 4096*512
    int row = i >> 9, e = i & 511;
    dst[i] = (f16_t)emb[(size_t)inp[row] * EME + e];
}

// Decoder GRU pointwise (EMD=768). gi/gh strides 3*EMD per batch row.
__global__ void gru_dec_kernel(const float* __restrict__ gi, const float* __restrict__ gh,
                               float* __restrict__ h32, f16_t* __restrict__ h16) {
    int i = blockIdx.x * 256 + threadIdx.x;      // 64*768
    int b = i / EMD, j = i % EMD;
    const float* gib = gi + b * 3 * EMD;
    const float* ghb = gh + b * 3 * EMD;
    float r = sigmoidf_(gib[j]           + ghb[j]);
    float z = sigmoidf_(gib[EMD + j]     + ghb[EMD + j]);
    float n = tanhf(    gib[2 * EMD + j] + r * ghb[2 * EMD + j]);
    float hold = h32[b * EMD + j];
    float h = (1.0f - z) * n + z * hold;
    h32[b * EMD + j] = h;
    h16[b * EMD + j] = (f16_t)h;
}

// hcat[(l*64+b)*512 + d*256 + j] = h32[((l*2+d)*64 + b)*256 + j]
__global__ void hcat_kernel(f16_t* dst, const float* h32) {
    int i = blockIdx.x * 256 + threadIdx.x;      // 2*64*512
    int l = i >> 15;
    int r = i & 32767;
    int b = r >> 9, c = r & 511;
    int d = c >> 8, j = c & 255;
    dst[i] = (f16_t)h32[(((l * 2 + d) * 64) + b) * 256 + j];
}

// al[s,b] = sum_e tanh(w1e[s*64+b,e] + w2h[b,e]) * Vp[e]   (one wave per (s,b))
__global__ void attn_logits_kernel(const float* __restrict__ w1e, const float* __restrict__ w2h,
                                   const float* __restrict__ Vp, float* __restrict__ al) {
    int lane = threadIdx.x & 31, wave = threadIdx.x >> 5;
    int pair = blockIdx.x * 8 + wave;            // 0..4095
    int s = pair >> 6, b = pair & 63;
    const float* we = w1e + (size_t)(s * 64 + b) * EMD;
    const float* wh = w2h + (size_t)b * EMD;
    float acc = 0.0f;
    for (int e = lane; e < EMD; e += 32) acc += tanhf(we[e] + wh[e]) * Vp[e];
    for (int m = 16; m > 0; m >>= 1) acc += __shfl_xor(acc, m, 32);
    if (lane == 0) al[s * 64 + b] = acc;
}

// softmax over s (axis 0) for each batch column b
__global__ void softmax_s_kernel(float* al) {
    int b = blockIdx.x, s = threadIdx.x;         // 64 threads
    __shared__ float sd[64];
    float v = al[s * 64 + b];
    sd[s] = v; __syncthreads();
    for (int m = 32; m > 0; m >>= 1) { if (s < m) sd[s] = fmaxf(sd[s], sd[s + m]); __syncthreads(); }
    float mx = sd[0]; __syncthreads();
    float e = expf(v - mx);
    sd[s] = e; __syncthreads();
    for (int m = 32; m > 0; m >>= 1) { if (s < m) sd[s] += sd[s + m]; __syncthreads(); }
    al[s * 64 + b] = e / sd[0];
}

// Xa[b,e] = sum_s a[s,b]*enc_out[s,b,e]; written into wgt_in f16 cols 768..1279
__global__ void xa_kernel(const float* __restrict__ al, const float* __restrict__ enc_out,
                          f16_t* __restrict__ wgt_in) {
    int i = blockIdx.x * 256 + threadIdx.x;      // 64*512
    int b = i >> 9, e = i & 511;
    float acc = 0.0f;
    for (int s = 0; s < SL; ++s) acc += al[s * 64 + b] * enc_out[(size_t)(s * 64 + b) * 512 + e];
    wgt_in[b * 1280 + 768 + e] = (f16_t)acc;
}

// wgt_in cols 0..767 = f16 emb_dec[dec_t[b]]
__global__ void dec_embed_kernel(f16_t* wgt_in, const f16_t* embdec16, const int* y, int t) {
    int b = blockIdx.x, tid = threadIdx.x;
    int tok = (t == 0) ? 0 : y[(t - 1) * 64 + b];
    const f16_t* src = embdec16 + (size_t)tok * EMD;
    for (int j = tid; j < EMD; j += 256) wgt_in[b * 1280 + j] = src[j];
}

// logits[b, 32000+s] = tanh( dot(sb_tanh[s,b,:], dec_out[b,:]) )
__global__ void score_c_kernel(const float* __restrict__ sb, const float* __restrict__ h1,
                               float* __restrict__ logits) {
    int lane = threadIdx.x & 31, wave = threadIdx.x >> 5;
    int pair = blockIdx.x * 8 + wave;            // 0..4095
    int b = pair >> 6, s = pair & 63;
    const float* sp = sb + (size_t)(s * 64 + b) * EMD;
    const float* hp = h1 + (size_t)b * EMD;
    float acc = 0.0f;
    for (int e = lane; e < EMD; e += 32) acc += sp[e] * hp[e];
    for (int m = 16; m > 0; m >>= 1) acc += __shfl_xor(acc, m, 32);
    if (lane == 0) logits[(size_t)b * VC + VSZ + s] = tanhf(acc);
}

// in-place softmax over the 32064-wide row for each b
__global__ void softmax_row_kernel(float* logits) {
    int b = blockIdx.x, tid = threadIdx.x;       // 256 threads
    float* row = logits + (size_t)b * VC;
    __shared__ float sd[256];
    float mx = -1e30f;
    for (int j = tid; j < VC; j += 256) mx = fmaxf(mx, row[j]);
    sd[tid] = mx; __syncthreads();
    for (int m = 128; m > 0; m >>= 1) { if (tid < m) sd[tid] = fmaxf(sd[tid], sd[tid + m]); __syncthreads(); }
    mx = sd[0]; __syncthreads();
    float sum = 0.0f;
    for (int j = tid; j < VC; j += 256) { float e = expf(row[j] - mx); row[j] = e; sum += e; }
    sd[tid] = sum; __syncthreads();
    for (int m = 128; m > 0; m >>= 1) { if (tid < m) sd[tid] += sd[tid + m]; __syncthreads(); }
    float inv = 1.0f / sd[0];
    __syncthreads();
    for (int j = tid; j < VC; j += 256) row[j] *= inv;
}

// res_t[b,j] = (j<32000) ? probs[b,j] : 0
__global__ void out_copy_kernel(float* __restrict__ res_t, const float* __restrict__ probs) {
    int i = blockIdx.x * 256 + threadIdx.x;
    if (i >= BS * VC) return;
    int j = i % VC;
    res_t[i] = (j < VSZ) ? probs[i] : 0.0f;
}

// copy mechanism: res_t[b, y[s,b]] += probs[b, 32000+s]
__global__ void scatter_kernel(float* __restrict__ res_t, const float* __restrict__ probs,
                               const int* __restrict__ y) {
    int i = blockIdx.x * 256 + threadIdx.x;      // 4096
    int s = i >> 6, b = i & 63;
    int tok = y[s * 64 + b];
    atomicAdd(res_t + (size_t)b * VC + tok, probs[(size_t)b * VC + VSZ + s]);
}

// ---------------------------------------------------------------------------
// Host orchestration (all launches on `stream`; graph-capture safe)
// ---------------------------------------------------------------------------
extern "C" void kernel_launch(void* const* d_in, const int* in_sizes, int n_in,
                              void* d_out, int out_size, void* d_ws, size_t ws_size,
                              hipStream_t stream) {
    (void)in_sizes; (void)n_in; (void)out_size; (void)ws_size;

    const int*   inp      = (const int*)  d_in[0];
    const int*   yin      = (const int*)  d_in[1];
    const float* emb_enc  = (const float*)d_in[2];
    const float* enc_Wih  = (const float*)d_in[3];
    const float* enc_Whh  = (const float*)d_in[4];
    const float* enc_bih  = (const float*)d_in[5];
    const float* enc_bhh  = (const float*)d_in[6];
    const float* out_encW = (const float*)d_in[7];
    const float* emb_dec  = (const float*)d_in[8];
    const float* dec_Wih  = (const float*)d_in[9];
    const float* dec_Whh  = (const float*)d_in[10];
    const float* dec_bih  = (const float*)d_in[11];
    const float* dec_bhh  = (const float*)d_in[12];
    const float* out_b    = (const float*)d_in[13];
    const float* W1       = (const float*)d_in[14];
    const float* l2_W     = (const float*)d_in[15];
    const float* l2_b     = (const float*)d_in[16];
    const float* l3_W     = (const float*)d_in[17];
    const float* l3_b     = (const float*)d_in[18];
    const float* Vp       = (const float*)d_in[19];

    // bump allocator over d_ws
    char* base = (char*)d_ws;
    size_t pos = 0;
    auto alloc = [&](size_t bytes) -> void* {
        void* p = base + pos;
        pos = (pos + bytes + 255) & ~(size_t)255;
        return p;
    };

    f16_t* encWih16 = (f16_t*)alloc((size_t)2*2*768*512*2);
    f16_t* encWhh16 = (f16_t*)alloc((size_t)2*2*768*256*2);
    f16_t* outW16   = (f16_t*)alloc((size_t)768*512*2);
    f16_t* W1T16    = (f16_t*)alloc((size_t)768*512*2);
    f16_t* l2W16    = (f16_t*)alloc((size_t)768*768*2);
    f16_t* l3W16    = (f16_t*)alloc((size_t)768*1280*2);
    f16_t* decWih16 = (f16_t*)alloc((size_t)2*2304*768*2);
    f16_t* decWhh16 = (f16_t*)alloc((size_t)2*2304*768*2);
    f16_t* embdec16 = (f16_t*)alloc((size_t)VSZ*EMD*2);
    f16_t* embx16   = (f16_t*)alloc((size_t)4096*512*2);
    float* giF      = (float*)alloc((size_t)4096*768*4);
    float* giB      = (float*)alloc((size_t)4096*768*4);
    float* h32      = (float*)alloc((size_t)2*2*64*256*4);
    f16_t* lo0_16   = (f16_t*)alloc((size_t)4096*512*2);
    float* lo1_32   = (float*)alloc((size_t)4096*512*4);
    f16_t* lo1_16   = (f16_t*)alloc((size_t)4096*512*2);
    f16_t* hcat16   = (f16_t*)alloc((size_t)128*512*2);
    float* hdec32   = (float*)alloc((size_t)128*768*4);
    f16_t* hdec16   = (f16_t*)alloc((size_t)128*768*2);
    float* w1e      = (float*)alloc((size_t)4096*768*4);
    float* sbt      = (float*)alloc((size_t)4096*768*4);
    float* w2h      = (float*)alloc((size_t)64*768*4);
    float* al       = (float*)alloc((size_t)64*64*4);
    f16_t* wgtIn16  = (f16_t*)alloc((size_t)64*1280*2);
    float* wgt32    = (float*)alloc((size_t)64*768*4);
    f16_t* wgt16    = (f16_t*)alloc((size_t)64*768*2);
    float* giDec    = (float*)alloc((size_t)64*2304*4);
    float* ghDec    = (float*)alloc((size_t)64*2304*4);
    float* logits   = (float*)alloc((size_t)64*VC*4);

    auto cvt = [&](f16_t* dst, const float* src, int n) {
        cvt_f16_kernel<<<(n + 255) / 256, 256, 0, stream>>>(dst, src, n);
    };
    // C[M,N] = act(A[M,K] @ B[N,K]^T + bias), optional f16 mirror
    auto gemm = [&](const f16_t* A, int lda, const f16_t* B, int ldb,
                    float* C, int ldc, f16_t* C16, int ldc16,
                    const float* bias, int M, int N, int K, int act) {
        dim3 g(N / 128, M / 16);
        gemm_nt_kernel<<<g, 256, 0, stream>>>(A, lda, B, ldb, C, ldc, C16, ldc16, bias, K, act);
    };

    // ---- weight conversion (once) ----
    cvt(encWih16, enc_Wih, 2*2*768*512);
    cvt(encWhh16, enc_Whh, 2*2*768*256);
    cvt(outW16,   out_encW, 768*512);
    cvt(l2W16,    l2_W,     768*768);
    cvt(l3W16,    l3_W,     768*1280);
    cvt(decWih16, dec_Wih,  2*2304*768);
    cvt(decWhh16, dec_Whh,  2*2304*768);
    cvt(embdec16, emb_dec,  VSZ*EMD);
    cvtT_f16_kernel<<<(768*512 + 255) / 256, 256, 0, stream>>>(W1T16, W1, 768, 512);

    // ---- encoder ----
    embed_enc_kernel<<<(4096*512) / 256, 256, 0, stream>>>(embx16, emb_enc, inp);

    const f16_t* layerIn = embx16;
    for (int l = 0; l < 2; ++l) {
        // batched input projections for both directions: (4096,512)@(512,768)
        for (int d = 0; d < 2; ++d) {
            gemm(layerIn, 512, encWih16 + (size_t)(l*2+d)*768*512, 512,
                 (d == 0 ? giF : giB), 768, nullptr, 0,
                 enc_bih + (l*2+d)*768, 4096, 768, 512, 0);
        }
        // fused persistent recurrence: one workgroup per direction
        gru_dir_fused_kernel<<<2, 1024, FUSED_LDS_BYTES, stream>>>(
            giF, giB,
            encWhh16 + (size_t)l*2*768*256,
            enc_bhh + l*2*768,
            h32 + (size_t)l*2*64*256,
            (l == 1) ? lo1_32 : nullptr,
            (l == 0) ? lo0_16 : lo1_16);
        layerIn = lo0_16;
    }

    // ---- encoder epilogue ----
    hcat_kernel<<<(2*64*512) / 256, 256, 0, stream>>>(hcat16, h32);
    // h = stack(h_fin) @ out_enc_W.T : (128,512)@(512,768), mirrored to f16
    gemm(hcat16, 512, outW16, 512, hdec32, 768, hdec16, 768, nullptr, 128, 768, 512, 0);
    // w1e = enc_out @ W1 : (4096,512)@(512,768)
    gemm(lo1_16, 512, W1T16, 512, w1e, 768, nullptr, 0, nullptr, 4096, 768, 512, 0);
    // sc_base (stored (s,b,e)) = tanh(enc_out @ out_enc_W.T)
    gemm(lo1_16, 512, outW16, 512, sbt, 768, nullptr, 0, nullptr, 4096, 768, 512, 1);

    // ---- decoder: 48 sequential steps ----
    for (int t = 0; t < OUT_SL; ++t) {
        float* res_t = (float*)d_out + (size_t)t * BS * VC;

        // w2h = h[1] @ l2_W.T + l2_b
        gemm(hdec16 + (size_t)64*768, 768, l2W16, 768, w2h, 768, nullptr, 0,
             l2_b, 64, 768, 768, 0);
        attn_logits_kernel<<<512, 256, 0, stream>>>(w1e, w2h, Vp, al);
        softmax_s_kernel<<<64, 64, 0, stream>>>(al);
        xa_kernel<<<(64*512) / 256, 256, 0, stream>>>(al, lo1_32, wgtIn16);
        dec_embed_kernel<<<64, 256, 0, stream>>>(wgtIn16, embdec16, yin, t);

        // wgt = concat(e, Xa) @ l3_W.T + l3_b : (64,1280)@(1280,768)
        gemm(wgtIn16, 1280, l3W16, 1280, wgt32, 768, wgt16, 768, l3_b, 64, 768, 1280, 0);

        // decoder GRU layer 0
        gemm(wgt16, 768, decWih16, 768, giDec, 2304, nullptr, 0, dec_bih, 64, 2304, 768, 0);
        gemm(hdec16, 768, decWhh16, 768, ghDec, 2304, nullptr, 0, dec_bhh, 64, 2304, 768, 0);
        gru_dec_kernel<<<(64*768) / 256, 256, 0, stream>>>(giDec, ghDec, hdec32, hdec16);

        // decoder GRU layer 1 (input = new h0)
        gemm(hdec16, 768, decWih16 + (size_t)2304*768, 768, giDec, 2304, nullptr, 0,
             dec_bih + 2304, 64, 2304, 768, 0);
        gemm(hdec16 + (size_t)64*768, 768, decWhh16 + (size_t)2304*768, 768, ghDec, 2304,
             nullptr, 0, dec_bhh + 2304, 64, 2304, 768, 0);
        gru_dec_kernel<<<(64*768) / 256, 256, 0, stream>>>(
            giDec, ghDec, hdec32 + (size_t)64*768, hdec16 + (size_t)64*768);

        // outp = dec_out @ emb_dec.T + out_b : (64,768)@(768,32000), ldc=32064
        gemm(hdec16 + (size_t)64*768, 768, embdec16, 768, logits, VC, nullptr, 0,
             out_b, 64, VSZ, 768, 0);
        score_c_kernel<<<512, 256, 0, stream>>>(sbt, hdec32 + (size_t)64*768, logits);
        softmax_row_kernel<<<64, 256, 0, stream>>>(logits);

        out_copy_kernel<<<(BS*VC + 255) / 256, 256, 0, stream>>>(res_t, logits);
        scatter_kernel<<<(SL*BS) / 256, 256, 0, stream>>>(res_t, logits, yin);
    }
}
